// AtlasV4Enhanced_29308856828245
// MI455X (gfx1250) — matrix operand order repaired and checked by reading, not compiled, not run on hardware
//
#include <hip/hip_runtime.h>
#include <hip/hip_bf16.h>
#include <math.h>

#define BATCH  8
#define SEQ    900
#define DMODEL 512
#define NHEAD  8
#define DKH    64
#define FFD    2048
#define SPAD   960              // SEQ padded to multiple of 64 for WMMA K-steps
#define MTOT   (BATCH * SEQ)    // 7200
#define LDSPITCH 72             // B-panel LDS row pitch (elems): 144B = 36 dwords
                                // -> (36*lane) mod 64 is a perfect bank permutation

typedef __attribute__((ext_vector_type(16))) __bf16          v16bf;
typedef __attribute__((ext_vector_type(8)))  float           v8f;
typedef __attribute__((ext_vector_type(8)))  unsigned short  us8;

union FragU { v16bf v; us8 h[2]; };

__device__ __forceinline__ unsigned short f2bf(float f) {
  unsigned int u = __float_as_uint(f);
  u += 0x7FFFu + ((u >> 16) & 1u);           // round-to-nearest-even
  return (unsigned short)(u >> 16);
}

// Issue one 16-byte async global->LDS copy (CDNA5, tracked by ASYNCcnt).
__device__ __forceinline__ void async_copy_b128(unsigned lds_off,
                                                unsigned long long gaddr) {
  asm volatile("global_load_async_to_lds_b128 %0, %1, off"
               :: "v"(lds_off), "v"(gaddr) : "memory");
}
__device__ __forceinline__ void wait_async0() {
  asm volatile("s_wait_asynccnt 0x0" ::: "memory");
}

// ---------------------------------------------------------------------------
// bf16 WMMA GEMM:  C[m,n] = sum_k A[m,k] * Bw[n,k]   (Bw pre-transposed [N,K])
// Workgroup = 8 waves. Wave w -> rows [blockM*128*RM + w*16*RM, +16*RM),
// cols [blockN*64, +64). B panel (64 rows x 64-K chunk) staged in LDS with a
// bank-conflict-free padded pitch, double-buffered via async global->LDS
// copies; A streamed from global. K must be a multiple of 64.
// MODE 0: (+bias?) -> f32    MODE 1: *scale + spatial bias -> f32
// MODE 2: +bias, exact GELU -> bf16    MODE 3: +bias +residual -> f32
// ---------------------------------------------------------------------------
template <int MODE, int RM>
__global__ __launch_bounds__(256) void gemm_wmma(
    const unsigned short* __restrict__ A,  long sAz, int lda,
    const unsigned short* __restrict__ Bw, long sBz, int ldb,
    float* __restrict__ Cf, unsigned short* __restrict__ Cb, long sCz, int ldc,
    int M, int N, int K,
    const float* __restrict__ bias,
    const float* __restrict__ res,
    const unsigned short* __restrict__ idx2,
    const float* __restrict__ comb,
    float scale)
{
  __shared__ __align__(16) unsigned short ldsB[2][64 * LDSPITCH]; // 2 x 9 KB

  const int tid  = threadIdx.x;
  const int lane = tid & 31;
  const int wave = tid >> 5;
  const int z    = blockIdx.z;
  A  += (long)z * sAz;
  Bw += (long)z * sBz;

  const int mBase = blockIdx.x * (128 * RM) + wave * (16 * RM);
  const int nBase = blockIdx.y * 64;

  const int kbA = (lane >> 4) << 3;         // 0 or 8  (16-bit A layout)
  const int kbB = (lane >> 4) << 3;         // 0 or 8  (16-bit B layout)
  const int nLane = lane & 15;

  const unsigned short* aRow[RM];
#pragma unroll
  for (int mt = 0; mt < RM; ++mt) {
    int mA = mBase + mt * 16 + nLane;
    if (mA >= M) mA = M - 1;                // clamp; stores are guarded
    aRow[mt] = A + (long)mA * lda;
  }

  // --- cooperative B staging: thread covers two 8-element groups per chunk ---
  const int g0 = tid, g1 = tid + 256;       // 512 groups = 64 rows x 8 groups
  const int row0 = g0 >> 3, c0 = (g0 & 7) * 8;
  const int row1 = g1 >> 3, c1 = (g1 & 7) * 8;
  int gr0 = nBase + row0; if (gr0 >= N) gr0 = N - 1;
  int gr1 = nBase + row1; if (gr1 >= N) gr1 = N - 1;
  const unsigned short* gp0 = Bw + (long)gr0 * ldb + c0;
  const unsigned short* gp1 = Bw + (long)gr1 * ldb + c1;
  const unsigned d0 = (unsigned)(size_t)&ldsB[0][row0 * LDSPITCH + c0];
  const unsigned d1 = (unsigned)(size_t)&ldsB[0][row1 * LDSPITCH + c1];
  const unsigned bufStride = 64 * LDSPITCH * 2;   // bytes

  auto issueChunk = [&](int k0, int buf) {
    async_copy_b128(d0 + buf * bufStride, (unsigned long long)(gp0 + k0));
    async_copy_b128(d1 + buf * bufStride, (unsigned long long)(gp1 + k0));
  };

  v8f acc[RM][4];
#pragma unroll
  for (int mt = 0; mt < RM; ++mt)
#pragma unroll
    for (int t = 0; t < 4; ++t) acc[mt][t] = (v8f){};

  issueChunk(0, 0);
  wait_async0();
  __syncthreads();

  for (int k0 = 0; k0 < K; k0 += 64) {
    const int  buf  = (k0 >> 6) & 1;
    const bool more = (k0 + 64) < K;
    if (more) issueChunk(k0 + 64, buf ^ 1);   // prefetch under compute

    // A fragments for both 32-deep halves (global, 16B loads)
    FragU af[2][RM];
#pragma unroll
    for (int h = 0; h < 2; ++h)
#pragma unroll
      for (int mt = 0; mt < RM; ++mt) {
        const unsigned short* ap = aRow[mt] + k0 + h * 32 + kbA;
        af[h][mt].h[0] = *(const us8*)(ap);
        af[h][mt].h[1] = *(const us8*)(ap + 16);
      }

    // B fragments from LDS panel (conflict-free padded pitch)
    const unsigned short* lb = &ldsB[buf][0];
    FragU bfr[2][4];
#pragma unroll
    for (int h = 0; h < 2; ++h)
#pragma unroll
      for (int t = 0; t < 4; ++t) {
        const unsigned short* bp =
            lb + (t * 16 + nLane) * LDSPITCH + h * 32 + kbB;
        bfr[h][t].h[0] = *(const us8*)(bp);
        bfr[h][t].h[1] = *(const us8*)(bp + 16);
      }

    // 16 (RM=2) back-to-back WMMAs
#pragma unroll
    for (int h = 0; h < 2; ++h)
#pragma unroll
      for (int mt = 0; mt < RM; ++mt)
#pragma unroll
        for (int t = 0; t < 4; ++t)
          acc[mt][t] = __builtin_amdgcn_wmma_f32_16x16x32_bf16(
              false, af[h][mt].v, false, bfr[h][t].v, (short)0,
              acc[mt][t], false, false);

    if (more) wait_async0();
    __syncthreads();
  }

  // --- epilogue ---
  const int mOff = (lane >> 4) << 3;        // C/D: lanes 16-31 hold M+8
#pragma unroll
  for (int mt = 0; mt < RM; ++mt) {
#pragma unroll
    for (int t = 0; t < 4; ++t) {
      const int n = nBase + t * 16 + nLane;
      if (n >= N) continue;
#pragma unroll
      for (int r = 0; r < 8; ++r) {
        const int m = mBase + mt * 16 + mOff + r;
        if (m >= M) continue;
        float c = acc[mt][t][r];
        if (MODE == 0) {
          if (bias) c += bias[n];
          Cf[(long)z * sCz + (long)m * ldc + n] = c;
        } else if (MODE == 1) {
          c = c * scale + comb[(int)idx2[(long)m * N + n] * NHEAD + z];
          Cf[(long)z * sCz + (long)m * ldc + n] = c;
        } else if (MODE == 2) {
          c += bias[n];
          float gl = 0.5f * c * (1.0f + erff(c * 0.70710678118654752f));
          Cb[(long)z * sCz + (long)m * ldc + n] = f2bf(gl);
        } else {
          c += bias[n] + res[(long)m * ldc + n];
          Cf[(long)z * sCz + (long)m * ldc + n] = c;
        }
      }
    }
  }
}

// ---------------------------------------------------------------------------
// Row softmax over 900 valid cols (row pitch SPAD); writes bf16 probs, zero pad.
// ---------------------------------------------------------------------------
__global__ __launch_bounds__(256) void softmax_kernel(
    const float* __restrict__ scores, unsigned short* __restrict__ probs)
{
  __shared__ float red[256];
  const int row = blockIdx.x;                    // h*SEQ + query
  const float* sp = scores + (long)row * SPAD;
  unsigned short* pp = probs + (long)row * SPAD;
  const int tid = threadIdx.x;

  float v[4];
#pragma unroll
  for (int it = 0; it < 4; ++it) {
    int j = tid + it * 256;
    v[it] = (j < SEQ) ? sp[j] : -3.0e38f;
  }
  float mx = fmaxf(fmaxf(v[0], v[1]), fmaxf(v[2], v[3]));
  red[tid] = mx; __syncthreads();
  for (int s = 128; s > 0; s >>= 1) {
    if (tid < s) red[tid] = fmaxf(red[tid], red[tid + s]);
    __syncthreads();
  }
  mx = red[0]; __syncthreads();

  float e[4]; float sum = 0.f;
#pragma unroll
  for (int it = 0; it < 4; ++it) {
    int j = tid + it * 256;
    e[it] = (j < SEQ) ? expf(v[it] - mx) : 0.f;
    sum += e[it];
  }
  red[tid] = sum; __syncthreads();
  for (int s = 128; s > 0; s >>= 1) {
    if (tid < s) red[tid] += red[tid + s];
    __syncthreads();
  }
  const float inv = 1.0f / red[0];

#pragma unroll
  for (int it = 0; it < 4; ++it) {
    int j = tid + it * 256;
    if (j < SPAD) pp[j] = (j < SEQ) ? f2bf(e[it] * inv) : (unsigned short)0;
  }
}

// ---------------------------------------------------------------------------
// LayerNorm over D=512 (256 threads x 2 elems); optional bf16 mirror output.
// ---------------------------------------------------------------------------
__global__ __launch_bounds__(256) void ln_kernel(
    const float* __restrict__ xin, const float* __restrict__ g,
    const float* __restrict__ bt, float* __restrict__ out,
    unsigned short* __restrict__ out_bf)
{
  __shared__ float r1[256], r2[256];
  const int row = blockIdx.x;
  const float* xp = xin + (long)row * DMODEL;
  const int tid = threadIdx.x;
  float a = xp[tid], b = xp[tid + 256];
  r1[tid] = a + b; r2[tid] = a * a + b * b;
  __syncthreads();
  for (int s = 128; s > 0; s >>= 1) {
    if (tid < s) { r1[tid] += r1[tid + s]; r2[tid] += r2[tid + s]; }
    __syncthreads();
  }
  const float mean = r1[0] * (1.0f / DMODEL);
  const float var  = r2[0] * (1.0f / DMODEL) - mean * mean;
  const float rstd = rsqrtf(var + 1e-5f);
  float y0 = (a - mean) * rstd * g[tid] + bt[tid];
  float y1 = (b - mean) * rstd * g[tid + 256] + bt[tid + 256];
  out[(long)row * DMODEL + tid] = y0;
  out[(long)row * DMODEL + tid + 256] = y1;
  if (out_bf) {
    out_bf[(long)row * DMODEL + tid] = f2bf(y0);
    out_bf[(long)row * DMODEL + tid + 256] = f2bf(y1);
  }
}

// --------------------------- small prep kernels ----------------------------
__global__ void wt_kernel(const float* __restrict__ w,
                          unsigned short* __restrict__ wt, int Kd, int Nd) {
  long i = (long)blockIdx.x * 256 + threadIdx.x;
  if (i >= (long)Kd * Nd) return;
  int k = (int)(i / Nd), n = (int)(i % Nd);
  wt[(long)n * Kd + k] = f2bf(w[i]);
}

__global__ void cvt_kernel(const float* __restrict__ in,
                           unsigned short* __restrict__ out, long n) {
  long i = (long)blockIdx.x * 256 + threadIdx.x;
  if (i < n) out[i] = f2bf(in[i]);
}

// f32 [B*S, D] -> bf16 [B, H, S, DK]
__global__ void pack_qk_kernel(const float* __restrict__ in,
                               unsigned short* __restrict__ out) {
  long i = (long)blockIdx.x * 256 + threadIdx.x;
  if (i >= (long)MTOT * DMODEL) return;
  int srow = (int)(i / DMODEL), c = (int)(i % DMODEL);
  int b = srow / SEQ, si = srow % SEQ, h = c / DKH, dk = c % DKH;
  out[(((long)(b * NHEAD + h) * SEQ) + si) * DKH + dk] = f2bf(in[i]);
}

// f32 [B*S, D] -> bf16 [B, H, DK, SPAD] (transposed, zero-padded in S)
__global__ void pack_vT_kernel(const float* __restrict__ in,
                               unsigned short* __restrict__ out) {
  long i = (long)blockIdx.x * 256 + threadIdx.x;
  const long total = (long)BATCH * NHEAD * DKH * SPAD;
  if (i >= total) return;
  int sp = (int)(i % SPAD); long r = i / SPAD;
  int dk = (int)(r % DKH);  r /= DKH;
  int h  = (int)(r % NHEAD);
  int b  = (int)(r / NHEAD);
  float v = (sp < SEQ) ? in[((long)(b * SEQ + sp)) * DMODEL + h * DKH + dk] : 0.f;
  out[i] = f2bf(v);
}

// packed spatial index table: idx = dist_idx*16 + dir_idx
__global__ void idx_kernel(unsigned short* __restrict__ idx2) {
  int i = blockIdx.x * 256 + threadIdx.x;
  if (i >= SEQ * SEQ) return;
  int p = i / SEQ, q = i % SEQ;                 // p = query, q = key
  int h1 = p / 30, w1 = p % 30, h2 = q / 30, w2 = q % 30;
  int dh = h2 - h1, dw = w2 - w1;
  float dist = sqrtf((float)(dh * dh + dw * dw));
  int di = (int)dist; if (di > 59) di = 59;
  int dir = 0;
  if (!(dh == 0 && dw == 0)) {
    float ang = atan2f((float)dh, (float)dw) + 3.14159265358979323846f;
    int t = (int)(ang * (16.0f / 6.28318530717958647692f));
    dir = t & 15;
  }
  idx2[i] = (unsigned short)(di * 16 + dir);
}

// comb[d][r][h] = dist_bias[d,h] + dir_bias[r,h]
__global__ void comb_kernel(const float* __restrict__ dist_bias,
                            const float* __restrict__ dir_bias,
                            float* __restrict__ comb) {
  int i = blockIdx.x * 256 + threadIdx.x;
  if (i >= 60 * 16 * NHEAD) return;
  int d = i / (16 * NHEAD);
  int r = (i / NHEAD) % 16;
  int h = i % NHEAD;
  comb[i] = dist_bias[d * NHEAD + h] + dir_bias[r * NHEAD + h];
}

// ---------------------------------------------------------------------------
extern "C" void kernel_launch(void* const* d_in, const int* in_sizes, int n_in,
                              void* d_out, int out_size, void* d_ws, size_t ws_size,
                              hipStream_t stream) {
  (void)in_sizes; (void)n_in; (void)out_size; (void)ws_size;
  const float* x         = (const float*)d_in[0];
  const float* w_q       = (const float*)d_in[1];
  const float* b_q       = (const float*)d_in[2];
  const float* w_k       = (const float*)d_in[3];
  const float* b_k       = (const float*)d_in[4];
  const float* w_v       = (const float*)d_in[5];
  const float* b_v       = (const float*)d_in[6];
  const float* w_o       = (const float*)d_in[7];
  const float* b_o       = (const float*)d_in[8];
  const float* dist_bias = (const float*)d_in[9];
  const float* dir_bias  = (const float*)d_in[10];
  const float* ln1_g     = (const float*)d_in[11];
  const float* ln1_b     = (const float*)d_in[12];
  const float* w1        = (const float*)d_in[13];
  const float* b1        = (const float*)d_in[14];
  const float* w2        = (const float*)d_in[15];
  const float* b2        = (const float*)d_in[16];
  const float* ln2_g     = (const float*)d_in[17];
  const float* ln2_b     = (const float*)d_in[18];
  float* out = (float*)d_out;

  char* wp = (char*)d_ws;
  auto alloc = [&](size_t bytes) -> char* {
    char* p = wp; wp += (bytes + 255) & ~(size_t)255; return p;
  };
  unsigned short* wtq    = (unsigned short*)alloc((size_t)DMODEL * DMODEL * 2);
  unsigned short* wtk    = (unsigned short*)alloc((size_t)DMODEL * DMODEL * 2);
  unsigned short* wtv    = (unsigned short*)alloc((size_t)DMODEL * DMODEL * 2);
  unsigned short* wto    = (unsigned short*)alloc((size_t)DMODEL * DMODEL * 2);
  unsigned short* wt1    = (unsigned short*)alloc((size_t)FFD * DMODEL * 2);   // [N=FF, K=D]
  unsigned short* wt2    = (unsigned short*)alloc((size_t)DMODEL * FFD * 2);   // [N=D, K=FF]
  unsigned short* x_bf   = (unsigned short*)alloc((size_t)MTOT * DMODEL * 2);
  unsigned short* q_bf   = (unsigned short*)alloc((size_t)BATCH * NHEAD * SEQ * DKH * 2);
  unsigned short* k_bf   = (unsigned short*)alloc((size_t)BATCH * NHEAD * SEQ * DKH * 2);
  unsigned short* vT_bf  = (unsigned short*)alloc((size_t)BATCH * NHEAD * DKH * SPAD * 2);
  unsigned short* idx2   = (unsigned short*)alloc((size_t)SEQ * SEQ * 2);
  float*          comb   = (float*)alloc((size_t)60 * 16 * NHEAD * 4);
  float*          scores = (float*)alloc((size_t)NHEAD * SEQ * SPAD * 4);
  unsigned short* probs  = (unsigned short*)alloc((size_t)NHEAD * SEQ * SPAD * 2);
  float*          ctxF   = (float*)alloc((size_t)MTOT * DMODEL * 4);
  unsigned short* ctx_bf = (unsigned short*)alloc((size_t)MTOT * DMODEL * 2);
  float*          attnF  = (float*)alloc((size_t)MTOT * DMODEL * 4);
  unsigned short* attn_bf= (unsigned short*)alloc((size_t)MTOT * DMODEL * 2);
  unsigned short* hid_bf = (unsigned short*)alloc((size_t)MTOT * FFD * 2);
  float*          tmpF   = (float*)alloc((size_t)MTOT * DMODEL * 4);

  auto grid1 = [](long n) { return dim3((unsigned)((n + 255) / 256)); };
  auto gemmGrid = [](int M, int N, int Z, int RM) {
    return dim3((M + 128 * RM - 1) / (128 * RM), (N + 63) / 64, Z);
  };
  const unsigned short* nil_u16 = nullptr;
  const float* nil_f = nullptr;

  // --- one-time prep ---
  wt_kernel<<<grid1((long)DMODEL * DMODEL), 256, 0, stream>>>(w_q, wtq, DMODEL, DMODEL);
  wt_kernel<<<grid1((long)DMODEL * DMODEL), 256, 0, stream>>>(w_k, wtk, DMODEL, DMODEL);
  wt_kernel<<<grid1((long)DMODEL * DMODEL), 256, 0, stream>>>(w_v, wtv, DMODEL, DMODEL);
  wt_kernel<<<grid1((long)DMODEL * DMODEL), 256, 0, stream>>>(w_o, wto, DMODEL, DMODEL);
  wt_kernel<<<grid1((long)DMODEL * FFD), 256, 0, stream>>>(w1, wt1, DMODEL, FFD);
  wt_kernel<<<grid1((long)FFD * DMODEL), 256, 0, stream>>>(w2, wt2, FFD, DMODEL);
  cvt_kernel<<<grid1((long)MTOT * DMODEL), 256, 0, stream>>>(x, x_bf, (long)MTOT * DMODEL);
  idx_kernel<<<grid1((long)SEQ * SEQ), 256, 0, stream>>>(idx2);
  comb_kernel<<<grid1(60 * 16 * NHEAD), 256, 0, stream>>>(dist_bias, dir_bias, comb);

  // --- QKV projections (bf16 WMMA), pack heads ---
  gemm_wmma<0, 2><<<gemmGrid(MTOT, DMODEL, 1, 2), 256, 0, stream>>>(
      x_bf, 0, DMODEL, wtq, 0, DMODEL, tmpF, nullptr, 0, DMODEL,
      MTOT, DMODEL, DMODEL, b_q, nil_f, nil_u16, nil_f, 0.f);
  pack_qk_kernel<<<grid1((long)MTOT * DMODEL), 256, 0, stream>>>(tmpF, q_bf);

  gemm_wmma<0, 2><<<gemmGrid(MTOT, DMODEL, 1, 2), 256, 0, stream>>>(
      x_bf, 0, DMODEL, wtk, 0, DMODEL, tmpF, nullptr, 0, DMODEL,
      MTOT, DMODEL, DMODEL, b_k, nil_f, nil_u16, nil_f, 0.f);
  pack_qk_kernel<<<grid1((long)MTOT * DMODEL), 256, 0, stream>>>(tmpF, k_bf);

  gemm_wmma<0, 2><<<gemmGrid(MTOT, DMODEL, 1, 2), 256, 0, stream>>>(
      x_bf, 0, DMODEL, wtv, 0, DMODEL, tmpF, nullptr, 0, DMODEL,
      MTOT, DMODEL, DMODEL, b_v, nil_f, nil_u16, nil_f, 0.f);
  pack_vT_kernel<<<grid1((long)BATCH * NHEAD * DKH * SPAD), 256, 0, stream>>>(tmpF, vT_bf);

  // --- attention: loop over batch, all heads in grid.z ---
  for (int b = 0; b < BATCH; ++b) {
    const unsigned short* qb = q_bf + (long)b * NHEAD * SEQ * DKH;
    const unsigned short* kb = k_bf + (long)b * NHEAD * SEQ * DKH;
    const unsigned short* vb = vT_bf + (long)b * NHEAD * DKH * SPAD;

    // scores = (Q Kt)/8 + spatial bias       [h, 900, 960(f32)]
    gemm_wmma<1, 2><<<gemmGrid(SEQ, SEQ, NHEAD, 2), 256, 0, stream>>>(
        qb, (long)SEQ * DKH, DKH, kb, (long)SEQ * DKH, DKH,
        scores, nullptr, (long)SEQ * SPAD, SPAD,
        SEQ, SEQ, DKH, nil_f, nil_f, idx2, comb, 0.125f);

    softmax_kernel<<<NHEAD * SEQ, 256, 0, stream>>>(scores, probs);

    // ctx = probs @ V   -> ctxF[b, s, h*64 + dk]
    gemm_wmma<0, 1><<<gemmGrid(SEQ, DKH, NHEAD, 1), 256, 0, stream>>>(
        probs, (long)SEQ * SPAD, SPAD, vb, (long)DKH * SPAD, SPAD,
        ctxF + (long)b * SEQ * DMODEL, nullptr, (long)DKH, DMODEL,
        SEQ, DKH, SPAD, nil_f, nil_f, nil_u16, nil_f, 0.f);
  }

  // --- output projection + residual, LN1 ---
  cvt_kernel<<<grid1((long)MTOT * DMODEL), 256, 0, stream>>>(ctxF, ctx_bf, (long)MTOT * DMODEL);
  gemm_wmma<3, 2><<<gemmGrid(MTOT, DMODEL, 1, 2), 256, 0, stream>>>(
      ctx_bf, 0, DMODEL, wto, 0, DMODEL, attnF, nullptr, 0, DMODEL,
      MTOT, DMODEL, DMODEL, b_o, x, nil_u16, nil_f, 0.f);
  ln_kernel<<<MTOT, 256, 0, stream>>>(attnF, ln1_g, ln1_b, attnF, attn_bf);

  // --- FFN: GEMM+GELU(bf16) -> GEMM+bias+residual -> LN2 ---
  gemm_wmma<2, 2><<<gemmGrid(MTOT, FFD, 1, 2), 256, 0, stream>>>(
      attn_bf, 0, DMODEL, wt1, 0, DMODEL, nullptr, hid_bf, 0, FFD,
      MTOT, FFD, DMODEL, b1, nil_f, nil_u16, nil_f, 0.f);
  gemm_wmma<3, 2><<<gemmGrid(MTOT, DMODEL, 1, 2), 256, 0, stream>>>(
      hid_bf, 0, FFD, wt2, 0, FFD, tmpF, nullptr, 0, DMODEL,
      MTOT, DMODEL, FFD, b2, attnF, nil_u16, nil_f, 0.f);
  ln_kernel<<<MTOT, 256, 0, stream>>>(tmpF, ln2_g, ln2_b, out, nullptr);
}
